// GatedGCNLayer_69784628625690
// MI455X (gfx1250) — compile-verified, matrix operand order checked by reading, not run on hardware
//
#include <hip/hip_runtime.h>

typedef __bf16  v16bf __attribute__((ext_vector_type(16)));
typedef __bf16  v8bf  __attribute__((ext_vector_type(8)));
typedef float   v8f   __attribute__((ext_vector_type(8)));
typedef float   v8fv  __attribute__((ext_vector_type(8)));

#define NN   40000
#define NE   640000
#define DD   128
#define NH   (NN * DD)      // 5,120,000
#define NEL  (NE * DD)      // 81,920,000
#define BN_EPS 1e-5f

// ---------------- helpers ----------------
static __device__ __forceinline__ __bf16 f32_to_bf16(float f) {
    return (__bf16)f;   // fptrunc, RNE -> native v_cvt on gfx1250
}

// load two 32B-aligned 8-float runs and convert to a 16-wide bf16 A fragment
static __device__ __forceinline__ v16bf load_a_frag(const float* __restrict__ rp,
                                                    int k0, int half) {
    const v8fv lo = *(const v8fv*)(rp + k0 + half * 8);
    const v8fv hi = *(const v8fv*)(rp + k0 + 16 + half * 8);
    const v8bf blo = __builtin_convertvector(lo, v8bf);
    const v8bf bhi = __builtin_convertvector(hi, v8bf);
    v16bf a;
    #pragma unroll
    for (int i = 0; i < 8; ++i) { a[i] = blo[i]; a[8 + i] = bhi[i]; }
    return a;
}

// ---------------- kernel 0: zero scratch ----------------
__global__ void k_zero(float* __restrict__ p, int n) {
    int i = blockIdx.x * blockDim.x + threadIdx.x;
    if (i < n) p[i] = 0.0f;
}

// ---------------- kernel 1: pack W (5 mats) f32 -> bf16 fragment layout ----
// layout: pack[((m*4 + kb)*128 + col)*32 + half*16 + slot]
//   slot 0..7  -> k = kb*32 + half*8 + slot
//   slot 8..15 -> k = kb*32 + 16 + half*8 + (slot-8)
__global__ void k_packW(const float* __restrict__ WA, const float* __restrict__ WB,
                        const float* __restrict__ WC, const float* __restrict__ WD,
                        const float* __restrict__ WE, __bf16* __restrict__ pack) {
    int i = blockIdx.x * blockDim.x + threadIdx.x;   // 5*4*128*32 = 81920
    if (i >= 5 * 4 * 128 * 32) return;
    int slot = i & 31;
    int col  = (i >> 5) & 127;
    int kb   = (i >> 12) & 3;
    int m    = i >> 14;
    int half = slot >> 4;
    int s    = slot & 15;
    int k    = kb * 32 + ((s < 8) ? (half * 8 + s) : (16 + half * 8 + (s - 8)));
    const float* W = (m == 0) ? WA : (m == 1) ? WB : (m == 2) ? WC : (m == 3) ? WD : WE;
    pack[i] = f32_to_bf16(W[k * DD + col]);
}

// ---------------- kernel 2: fused node GEMMs  Ah,Bh,Dh,Eh = h @ {WA,WB,WD,WE}
// block = 256 threads = 8 waves; block covers 32 rows x 128 cols for 4 mats.
// wave w: rows = blk*32 + (w>>2)*16, cols = (w&3)*32 (two 16-col tiles).
__global__ void __launch_bounds__(256)
k_node_gemm4(const float* __restrict__ h, const __bf16* __restrict__ pack,
             float* __restrict__ Ah, float* __restrict__ Bh,
             float* __restrict__ Dh, float* __restrict__ Eh) {
    const int wave = threadIdx.x >> 5;
    const int lane = threadIdx.x & 31;
    const int half = lane >> 4;
    const int r    = lane & 15;
    const int rowBase = blockIdx.x * 32 + (wave >> 2) * 16;
    const int n0      = (wave & 3) * 32;
    // pack slice indices for the 4 matrices we need: WA=0,WB=1,WD=3,WE=4
    const int mats[4] = {0, 1, 3, 4};

    v8f acc[4][2];
    #pragma unroll
    for (int m = 0; m < 4; ++m)
        #pragma unroll
        for (int t = 0; t < 2; ++t) acc[m][t] = (v8f)(0.0f);

    const float* hp = h + (size_t)(rowBase + r) * DD;

    #pragma unroll
    for (int kb = 0; kb < 4; ++kb) {
        const int k0 = kb * 32;
        const v16bf a = load_a_frag(hp, k0, half);

        #pragma unroll
        for (int m = 0; m < 4; ++m) {
            #pragma unroll
            for (int t = 0; t < 2; ++t) {
                const int col = n0 + t * 16 + r;
                const v16bf b = *(const v16bf*)(pack +
                    (((size_t)(mats[m] * 4 + kb) * 128 + col) * 32 + half * 16));
                acc[m][t] = __builtin_amdgcn_wmma_f32_16x16x32_bf16(
                    false, a, false, b, (short)0, acc[m][t], false, false);
            }
        }
    }

    float* outs[4] = {Ah, Bh, Dh, Eh};
    #pragma unroll
    for (int m = 0; m < 4; ++m)
        #pragma unroll
        for (int t = 0; t < 2; ++t)
            #pragma unroll
            for (int j = 0; j < 8; ++j)
                outs[m][(size_t)(rowBase + half * 8 + j) * DD + n0 + t * 16 + r] = acc[m][t][j];
}

// ---------------- kernel 3: edge GEMM + gate + scatter + graph-norm + e-stats
// Ce = e @ WC; g = relu(Ce + Dh[src] + Eh[dst]); sig = sigmoid(g)
// atomic num[dst] += sig*Bh[src]; den[dst] += sig; e_tmp = g*snorm_e -> d_out
// per-column sum / sumsq of e_tmp accumulated via LDS ds_add then global atomics
__global__ void __launch_bounds__(256)
k_edge_fused(const float* __restrict__ e, const __bf16* __restrict__ pack,
             const int* __restrict__ src, const int* __restrict__ dst,
             const float* __restrict__ snorm_e,
             const float* __restrict__ Bh, const float* __restrict__ Dh,
             const float* __restrict__ Eh,
             float* __restrict__ num, float* __restrict__ den,
             float* __restrict__ outE,
             float* __restrict__ gsum_e, float* __restrict__ gsq_e) {
    __shared__ float s_sum[DD];
    __shared__ float s_sq[DD];
    if (threadIdx.x < DD) { s_sum[threadIdx.x] = 0.0f; s_sq[threadIdx.x] = 0.0f; }
    __syncthreads();

    const int wave = threadIdx.x >> 5;
    const int lane = threadIdx.x & 31;
    const int half = lane >> 4;
    const int r    = lane & 15;
    const int eBase = blockIdx.x * 32 + (wave >> 2) * 16;
    const int n0    = (wave & 3) * 32;

    v8f acc[2];
    acc[0] = (v8f)(0.0f); acc[1] = (v8f)(0.0f);

    const float* ep = e + (size_t)(eBase + r) * DD;
    const __bf16* wc = pack + (size_t)(2 * 4) * 128 * 32;   // WC is mat index 2

    #pragma unroll
    for (int kb = 0; kb < 4; ++kb) {
        const int k0 = kb * 32;
        const v16bf a = load_a_frag(ep, k0, half);

        #pragma unroll
        for (int t = 0; t < 2; ++t) {
            const int col = n0 + t * 16 + r;
            const v16bf b = *(const v16bf*)(wc + (((size_t)kb * 128 + col) * 32 + half * 16));
            acc[t] = __builtin_amdgcn_wmma_f32_16x16x32_bf16(
                false, a, false, b, (short)0, acc[t], false, false);
        }
    }

    // epilogue: gather, gate, scatter-atomics, graph norm, stats
    int   sIdx[8], dIdx[8];
    float sn[8];
    #pragma unroll
    for (int j = 0; j < 8; ++j) {
        const int eI = eBase + half * 8 + j;
        sIdx[j] = src[eI];
        dIdx[j] = dst[eI];
        sn[j]   = snorm_e[eI];
    }

    #pragma unroll
    for (int t = 0; t < 2; ++t) {
        const int col = n0 + t * 16 + r;
        float psum = 0.0f, psq = 0.0f;
        #pragma unroll
        for (int j = 0; j < 8; ++j) {
            const int eI = eBase + half * 8 + j;
            float g = acc[t][j] + Dh[(size_t)sIdx[j] * DD + col]
                                + Eh[(size_t)dIdx[j] * DD + col];
            g = fmaxf(g, 0.0f);
            const float te = g * sn[j];
            outE[(size_t)eI * DD + col] = te;
            const float sig = 1.0f / (1.0f + __expf(-g));
            atomicAdd(&num[(size_t)dIdx[j] * DD + col],
                      sig * Bh[(size_t)sIdx[j] * DD + col]);
            atomicAdd(&den[(size_t)dIdx[j] * DD + col], sig);
            psum += te; psq += te * te;
        }
        atomicAdd(&s_sum[col], psum);
        atomicAdd(&s_sq[col],  psq);
    }
    __syncthreads();
    if (threadIdx.x < DD) {
        atomicAdd(&gsum_e[threadIdx.x], s_sum[threadIdx.x]);
        atomicAdd(&gsq_e[threadIdx.x],  s_sq[threadIdx.x]);
    }
}

// ---------------- kernel 4: node combine + graph norm + h-stats -------------
__global__ void __launch_bounds__(256)
k_node_combine(const float* __restrict__ Ah, const float* __restrict__ num,
               const float* __restrict__ den, const float* __restrict__ snorm_n,
               float* __restrict__ outH,
               float* __restrict__ gsum_h, float* __restrict__ gsq_h) {
    __shared__ float s_sum[DD];
    __shared__ float s_sq[DD];
    if (threadIdx.x < DD) { s_sum[threadIdx.x] = 0.0f; s_sq[threadIdx.x] = 0.0f; }
    __syncthreads();

    const int idx = blockIdx.x * blockDim.x + threadIdx.x;   // < NH exactly
    const int col = idx & (DD - 1);
    const int row = idx >> 7;
    const float t = (Ah[idx] + num[idx] / (den[idx] + 1e-6f)) * snorm_n[row];
    outH[idx] = t;
    atomicAdd(&s_sum[col], t);
    atomicAdd(&s_sq[col],  t * t);
    __syncthreads();
    if (threadIdx.x < DD) {
        atomicAdd(&gsum_h[threadIdx.x], s_sum[threadIdx.x]);
        atomicAdd(&gsq_h[threadIdx.x],  s_sq[threadIdx.x]);
    }
}

// ---------------- kernel 5: BN normalize + relu (in place on d_out) ---------
__global__ void __launch_bounds__(256)
k_bn_relu(float* __restrict__ x, const float* __restrict__ gsum,
          const float* __restrict__ gsq, const float* __restrict__ gamma,
          const float* __restrict__ beta, float invN, long long n) {
    const long long idx = (long long)blockIdx.x * blockDim.x + threadIdx.x;
    if (idx >= n) return;
    const int col = (int)(idx & (DD - 1));
    const float mu  = gsum[col] * invN;
    const float var = gsq[col] * invN - mu * mu;
    const float rs  = rsqrtf(var + BN_EPS);
    const float v   = (x[idx] - mu) * rs * gamma[col] + beta[col];
    x[idx] = fmaxf(v, 0.0f);
}

// ---------------- launcher --------------------------------------------------
extern "C" void kernel_launch(void* const* d_in, const int* in_sizes, int n_in,
                              void* d_out, int out_size, void* d_ws, size_t ws_size,
                              hipStream_t stream) {
    const float* h       = (const float*)d_in[0];
    const float* e       = (const float*)d_in[1];
    const int*   src     = (const int*)  d_in[2];
    const int*   dst     = (const int*)  d_in[3];
    const float* snorm_n = (const float*)d_in[4];
    const float* snorm_e = (const float*)d_in[5];
    const float* WA      = (const float*)d_in[6];
    const float* WB      = (const float*)d_in[7];
    const float* WC      = (const float*)d_in[8];
    const float* WD      = (const float*)d_in[9];
    const float* WE      = (const float*)d_in[10];
    const float* gamma_h = (const float*)d_in[11];
    const float* beta_h  = (const float*)d_in[12];
    const float* gamma_e = (const float*)d_in[13];
    const float* beta_e  = (const float*)d_in[14];

    float* ws = (float*)d_ws;
    float* Ah  = ws;
    float* Bh  = ws + (size_t)1 * NH;
    float* Dh  = ws + (size_t)2 * NH;
    float* Eh  = ws + (size_t)3 * NH;
    float* num = ws + (size_t)4 * NH;
    float* den = ws + (size_t)5 * NH;
    __bf16* pack = (__bf16*)(ws + (size_t)6 * NH);           // 5*128*128 bf16
    float* stats = (float*)(pack + 5 * 128 * 128);           // 4 x 128 floats
    float* gsum_h = stats;
    float* gsq_h  = stats + DD;
    float* gsum_e = stats + 2 * DD;
    float* gsq_e  = stats + 3 * DD;

    float* outH = (float*)d_out;                             // [NN, 128]
    float* outE = (float*)d_out + (size_t)NH;                // [NE, 128]

    // 0) zero num/den and stats
    k_zero<<<(2 * NH + 255) / 256, 256, 0, stream>>>(num, 2 * NH);
    k_zero<<<(4 * DD + 255) / 256, 256, 0, stream>>>(stats, 4 * DD);

    // 1) pack all five W matrices to bf16 fragment layout
    k_packW<<<(5 * 4 * 128 * 32 + 255) / 256, 256, 0, stream>>>(WA, WB, WC, WD, WE, pack);

    // 2) node GEMMs (Ah, Bh, Dh, Eh)
    k_node_gemm4<<<NN / 32, 256, 0, stream>>>(h, pack, Ah, Bh, Dh, Eh);

    // 3) edge GEMM fused with gate / sigmoid / scatter / graph-norm / stats
    k_edge_fused<<<NE / 32, 256, 0, stream>>>(e, pack, src, dst, snorm_e,
                                              Bh, Dh, Eh, num, den, outE,
                                              gsum_e, gsq_e);

    // 4) node combine + graph norm + stats
    k_node_combine<<<NH / 256, 256, 0, stream>>>(Ah, num, den, snorm_n,
                                                 outH, gsum_h, gsq_h);

    // 5) batch-norm + relu, in place on d_out
    k_bn_relu<<<(NH + 255) / 256, 256, 0, stream>>>(outH, gsum_h, gsq_h,
                                                    gamma_h, beta_h,
                                                    1.0f / NN, (long long)NH);
    k_bn_relu<<<(int)(((long long)NEL + 255) / 256), 256, 0, stream>>>(
        outE, gsum_e, gsq_e, gamma_e, beta_e, 1.0f / NE, (long long)NEL);
}
